// GNNModel_47115791237145
// MI455X (gfx1250) — matrix operand order repaired and checked by reading, not compiled, not run on hardware
//
#include <hip/hip_runtime.h>
#include <hip/hip_bf16.h>

typedef _Float16 v16h __attribute__((ext_vector_type(16)));
typedef float    v8f  __attribute__((ext_vector_type(8)));

#define D 64

// ---------------------------------------------------------------------------
// deg[i] = 1.0 (self-loop weight)
__global__ void init_deg_kernel(float* __restrict__ deg, int n) {
    int i = blockIdx.x * blockDim.x + threadIdx.x;
    if (i < n) deg[i] = 1.0f;
}

// deg[dst[e]] += w[e]
__global__ void deg_edges_kernel(const long long* __restrict__ dst,
                                 const float* __restrict__ w,
                                 float* __restrict__ deg, int E) {
    int e = blockIdx.x * blockDim.x + threadIdx.x;
    if (e >= E) return;
    int d = (int)dst[e];
    __hip_atomic_fetch_add(&deg[d], w[e], __ATOMIC_RELAXED, __HIP_MEMORY_SCOPE_AGENT);
}

// dinv[i] = deg>0 ? rsqrt(deg) : 0   (in place)
__global__ void dinv_kernel(float* __restrict__ deg, int n) {
    int i = blockIdx.x * blockDim.x + threadIdx.x;
    if (i >= n) return;
    float d = deg[i];
    deg[i] = (d > 0.0f) ? rsqrtf(d) : 0.0f;
}

// ---------------------------------------------------------------------------
// H = X @ W  where X is [n x 64] f32, W is [64 x 64] f32, H is [n x 64] f32.
// One wave per 16-row tile.  f16 operands, f32 accumulate via
// v_wmma_f32_16x16x32_f16 (2 K-steps of 32, 4 N-tiles of 16) = 8 WMMA/wave.
__global__ void gemm_wmma_kernel(const float* __restrict__ X,
                                 const float* __restrict__ W,
                                 float* __restrict__ H, int n) {
    int wave = (int)((blockIdx.x * blockDim.x + threadIdx.x) >> 5);
    int lane = threadIdx.x & 31;
    int ntiles = n >> 4;
    if (wave >= ntiles) return;   // whole wave exits together (EXEC all-1 for WMMA)

    int m  = lane & 15;   // row within A tile / column within B & C tiles
    int kh = lane >> 4;   // half-select (0 or 1)

    // B fragments: W[k][nn], k striped per 16-bit B layout.
    v16h bfrag[2][4];
#pragma unroll
    for (int kb = 0; kb < 2; ++kb) {
#pragma unroll
        for (int nt = 0; nt < 4; ++nt) {
            const float* wp = W + (size_t)(kb * 32 + kh * 8) * D + nt * 16 + m;
#pragma unroll
            for (int i = 0; i < 16; ++i)
                bfrag[kb][nt][i] = (_Float16)wp[(size_t)(i + (i & 8)) * D];
        }
    }

    v8f acc[4] = {};
    int row0 = wave << 4;
#pragma unroll
    for (int kb = 0; kb < 2; ++kb) {
        // A fragment: documented 16-bit 16x32 layout.
        // lane(kh=0): K {0..7,16..23}, lane(kh=1): K {8..15,24..31}
        const float* xrow = X + (size_t)(row0 + m) * D + kb * 32;
        v16h a;
#pragma unroll
        for (int i = 0; i < 8; ++i) {
            a[i]     = (_Float16)xrow[kh * 8 + i];
            a[8 + i] = (_Float16)xrow[16 + kh * 8 + i];
        }
#pragma unroll
        for (int nt = 0; nt < 4; ++nt)
            acc[nt] = __builtin_amdgcn_wmma_f32_16x16x32_f16(
                false, a, false, bfrag[kb][nt], (short)0, acc[nt], false, false);
    }

    // C/D layout: VGPR r, lanes 0-15 -> M=r, lanes 16-31 -> M=r+8, N=lane%16
#pragma unroll
    for (int nt = 0; nt < 4; ++nt) {
#pragma unroll
        for (int r = 0; r < 8; ++r)
            H[(size_t)(row0 + r + 8 * kh) * D + nt * 16 + m] = acc[nt][r];
    }
}

// ---------------------------------------------------------------------------
// A[i,:] = dinv[i]^2 * H[i,:]   (self-loop term initializes the accumulator)
__global__ void selfloop_init_kernel(const float* __restrict__ H,
                                     const float* __restrict__ dinv,
                                     float* __restrict__ A, int n) {
    int t = blockIdx.x * blockDim.x + threadIdx.x;
    int i = t >> 6, k = t & 63;
    if (i >= n) return;
    float di = dinv[i];
    A[(size_t)i * D + k] = di * di * H[(size_t)i * D + k];
}

// Per edge: A[dst,:] += (dinv[src]*w*dinv[dst]) * H[src,:]
// One wave per edge, 2 columns per lane, hardware f32 atomics.
__global__ void edge_scatter_kernel(const long long* __restrict__ src,
                                    const long long* __restrict__ dst,
                                    const float* __restrict__ w,
                                    const float* __restrict__ dinv,
                                    const float* __restrict__ H,
                                    float* __restrict__ A, int E) {
    long long t = (long long)blockIdx.x * blockDim.x + threadIdx.x;
    int e = (int)(t >> 5);
    int lane = (int)(t & 31);
    if (e >= E) return;
    int s = (int)src[e];
    int d = (int)dst[e];
    float nrm = dinv[s] * w[e] * dinv[d];
    float v0 = nrm * H[(size_t)s * D + lane];
    float v1 = nrm * H[(size_t)s * D + 32 + lane];
    __hip_atomic_fetch_add(&A[(size_t)d * D + lane], v0,
                           __ATOMIC_RELAXED, __HIP_MEMORY_SCOPE_AGENT);
    __hip_atomic_fetch_add(&A[(size_t)d * D + 32 + lane], v1,
                           __ATOMIC_RELAXED, __HIP_MEMORY_SCOPE_AGENT);
}

// A[i,k] = relu(A[i,k] + b[k])   (in place)
__global__ void bias_relu_kernel(float* __restrict__ A,
                                 const float* __restrict__ b, int n) {
    int t = blockIdx.x * blockDim.x + threadIdx.x;
    int i = t >> 6, k = t & 63;
    if (i >= n) return;
    float v = A[(size_t)i * D + k] + b[k];
    A[(size_t)i * D + k] = v > 0.0f ? v : 0.0f;
}

// out[i] = dot(H[i,:], Wfc) + bfc
__global__ void fc_out_kernel(const float* __restrict__ H,
                              const float* __restrict__ Wfc,
                              const float* __restrict__ bfc,
                              float* __restrict__ out, int n) {
    int i = blockIdx.x * blockDim.x + threadIdx.x;
    if (i >= n) return;
    const float4* h4 = (const float4*)(H + (size_t)i * D);
    const float4* w4 = (const float4*)Wfc;
    float s = 0.0f;
#pragma unroll
    for (int k = 0; k < 16; ++k) {
        float4 a = h4[k], b = w4[k];
        s += a.x * b.x + a.y * b.y + a.z * b.z + a.w * b.w;
    }
    out[i] = s + bfc[0];
}

// ---------------------------------------------------------------------------
extern "C" void kernel_launch(void* const* d_in, const int* in_sizes, int n_in,
                              void* d_out, int out_size, void* d_ws, size_t ws_size,
                              hipStream_t stream) {
    const float*     x   = (const float*)d_in[0];
    const long long* ei  = (const long long*)d_in[1];   // int64 [2,E]
    const float*     ew  = (const float*)d_in[2];
    const float*     W1  = (const float*)d_in[3];
    const float*     b1  = (const float*)d_in[4];
    const float*     W2  = (const float*)d_in[5];
    const float*     b2  = (const float*)d_in[6];
    const float*     Wfc = (const float*)d_in[7];
    const float*     bfc = (const float*)d_in[8];
    float*           out = (float*)d_out;

    const int n = in_sizes[0] / D;     // 50000 (multiple of 16)
    const int E = in_sizes[2];         // 800000
    const long long* src = ei;
    const long long* dst = ei + E;

    float* bufH = (float*)d_ws;                    // n*64  : X@W
    float* bufA = bufH + (size_t)n * D;            // n*64  : aggregation accumulator
    float* dinv = bufA + (size_t)n * D;            // n     : deg -> dinv (in place)

    const int BT = 256;
    dim3 blk(BT);
    int gN    = (n + BT - 1) / BT;                  // n threads
    int gE    = (E + BT - 1) / BT;                  // E threads
    int gNE   = (int)(((size_t)n * D + BT - 1) / BT);   // n*64 threads
    int gEdge = (int)(((size_t)E * 32 + BT - 1) / BT);  // E waves
    int gGemm = ((n / 16) * 32 + BT - 1) / BT;      // one wave per 16-row tile

    // ---- normalization ----
    init_deg_kernel<<<gN, blk, 0, stream>>>(dinv, n);
    deg_edges_kernel<<<gE, blk, 0, stream>>>(dst, ew, dinv, E);
    dinv_kernel<<<gN, blk, 0, stream>>>(dinv, n);

    // ---- layer 1 ----
    gemm_wmma_kernel<<<gGemm, blk, 0, stream>>>(x, W1, bufH, n);
    selfloop_init_kernel<<<gNE, blk, 0, stream>>>(bufH, dinv, bufA, n);
    edge_scatter_kernel<<<gEdge, blk, 0, stream>>>(src, dst, ew, dinv, bufH, bufA, E);
    bias_relu_kernel<<<gNE, blk, 0, stream>>>(bufA, b1, n);

    // ---- layer 2 ----
    gemm_wmma_kernel<<<gGemm, blk, 0, stream>>>(bufA, W2, bufH, n);
    selfloop_init_kernel<<<gNE, blk, 0, stream>>>(bufH, dinv, bufA, n);
    edge_scatter_kernel<<<gEdge, blk, 0, stream>>>(src, dst, ew, dinv, bufH, bufA, E);
    bias_relu_kernel<<<gNE, blk, 0, stream>>>(bufA, b2, n);

    // ---- readout ----
    fc_out_kernel<<<gN, blk, 0, stream>>>(bufA, Wfc, bfc, out, n);
}
